// LocalAttention_15393162789389
// MI455X (gfx1250) — matrix operand order repaired and checked by reading, not compiled, not run on hardware
//
#include <hip/hip_runtime.h>

typedef __attribute__((ext_vector_type(16))) _Float16 v16h;
typedef __attribute__((ext_vector_type(8)))  _Float16 v8h;
typedef __attribute__((ext_vector_type(8)))  float    v8f;
typedef __attribute__((ext_vector_type(4)))  unsigned int u32x4;
typedef __attribute__((ext_vector_type(8)))  int i32x8;
typedef __attribute__((ext_vector_type(4)))  int i32x4;

constexpr int Cd   = 256;   // channels
constexpr int Ld   = 1024;  // sequence length
constexpr int Wd   = 80;    // window
constexpr int Kp   = 96;    // window padded to multiple of 32 (WMMA K)
constexpr int Fd   = 256;   // feature dim

// ---------------- workspace byte offsets ----------------
constexpr size_t OFF_GT   = 0;        // f16 [96][96]
constexpr size_t OFF_AV   = 18432;    // f32 [96]
constexpr size_t OFF_BV   = 18816;    // f32 [96]
constexpr size_t OFF_S0   = 19200;    // f32 [1]
constexpr size_t OFF_WVWF = 19456;    // f32 [80][256]
constexpr size_t OFF_BVF  = 101376;   // f32 [256]

// =======================================================================
// Kernel 1: position-independent precompute (tiny, VALU only)
// =======================================================================
__global__ __launch_bounds__(256) void la_precompute(
    const float* __restrict__ Wq, const float* __restrict__ bq,
    const float* __restrict__ Wk, const float* __restrict__ bk,
    const float* __restrict__ Wv, const float* __restrict__ bv,
    const float* __restrict__ Wf, const float* __restrict__ bf,
    _Float16* __restrict__ Gt, float* __restrict__ avec,
    float* __restrict__ bvec, float* __restrict__ s0,
    float* __restrict__ WvWf, float* __restrict__ bvf)
{
  int idx = blockIdx.x * 256 + threadIdx.x;
  if (idx < Kp * Kp) {                       // Gt[n][k] = sum_f Wq[k,f]*Wk[n,f]
    int n = idx / Kp, k = idx - n * Kp;
    float acc = 0.f;
    if (n < Wd && k < Wd)
      for (int f = 0; f < Fd; ++f) acc += Wq[k*Fd+f] * Wk[n*Fd+f];
    Gt[idx] = (_Float16)acc;
    return;
  }
  idx -= Kp * Kp;
  if (idx < Kp) {
    float acc = 0.f;
    if (idx < Wd) for (int f = 0; f < Fd; ++f) acc += Wq[idx*Fd+f] * bk[f];
    avec[idx] = acc; return;
  }
  idx -= Kp;
  if (idx < Kp) {
    float acc = 0.f;
    if (idx < Wd) for (int f = 0; f < Fd; ++f) acc += Wk[idx*Fd+f] * bq[f];
    bvec[idx] = acc; return;
  }
  idx -= Kp;
  if (idx < 1) {
    float acc = 0.f;
    for (int f = 0; f < Fd; ++f) acc += bq[f] * bk[f];
    s0[0] = acc; return;
  }
  idx -= 1;
  if (idx < Wd * Cd) {                       // WvWf[j][c]
    int j = idx / Cd, c = idx - j * Cd;
    float acc = 0.f;
    for (int f = 0; f < Fd; ++f) acc += Wv[j*Fd+f] * Wf[f*Cd+c];
    WvWf[idx] = acc; return;
  }
  idx -= Wd * Cd;
  if (idx < Cd) {
    float acc = bf[idx];
    for (int f = 0; f < Fd; ++f) acc += bv[f] * Wf[f*Cd+idx];
    bvf[idx] = acc;
  }
}

// =======================================================================
// WMMA helpers (16x16x32 f16 -> f32), fragment loads per ISA VGPR layouts
// =======================================================================
__device__ __forceinline__ v8f wmma32(v16h a, v16h b, v8f c) {
  return __builtin_amdgcn_wmma_f32_16x16x32_f16(
      false, a, false, b, (short)0, c, false, false);
}

template <typename PTR>
__device__ __forceinline__ v16h loadA(PTR base, int stride, int m0, int k0, int lane) {
  int row = m0 + (lane & 15);
  int kb  = k0 + ((lane & 16) ? 8 : 0);
  const _Float16* p = base + row * stride + kb;
  v8h lo = *(const v8h*)(p);
  v8h hi = *(const v8h*)(p + 16);
  v16h a;
#pragma unroll
  for (int i = 0; i < 8; ++i) { a[i] = lo[i]; a[i + 8] = hi[i]; }
  return a;
}

template <typename PTR>
__device__ __forceinline__ v16h loadBt(PTR base, int stride, int n0, int k0, int lane) {
  int n  = n0 + (lane & 15);
  int kb = k0 + ((lane & 16) ? 8 : 0);
  const _Float16* p = base + n * stride + kb;
  v8h lo = *(const v8h*)(p);
  v8h hi = *(const v8h*)(p + 16);
  v16h b;
#pragma unroll
  for (int i = 0; i < 8; ++i) { b[i] = lo[i]; b[i + 8] = hi[i]; }
  return b;
}

// =======================================================================
// Kernel 2: one workgroup (8 wave32) per sequence position l
// =======================================================================
__global__ __launch_bounds__(256) void la_main(
    const float* __restrict__ x,
    const _Float16* __restrict__ Gt,
    const float* __restrict__ avec,
    const float* __restrict__ bvec,
    const float* __restrict__ s0p,
    const float* __restrict__ WvWf,
    const float* __restrict__ bvf,
    float* __restrict__ out)
{
  __shared__ _Float16 sWin[Cd][Kp];                  // 48 KB: A for H, B^T for S
  __shared__ _Float16 sH[Cd][Kp];                    // 48 KB: A for S
  __shared__ __align__(16) float sPool[Cd * Cd / 2]; // 128 KB: f32 TDM stage, then f16 S
  __shared__ float sWa[Cd], sWb[Cd];
  __shared__ float sRowMax[Cd], sRowInvZ[Cd];
  __shared__ float sPart[8][Cd];                     // column partial sums per wave
  __shared__ float sWcol[Cd];
  __shared__ float sU[Kp];

  float* sStage = sPool;                             // [256*80] f32 (TDM tile)
  _Float16 (*sS)[Cd] = (_Float16(*)[Cd])sPool;       // [256][256] f16 (scores)

  const int l    = blockIdx.x;
  const int tid  = threadIdx.x;
  const int lane = tid & 31;
  const int wave = tid >> 5;

  int start = l - (Wd / 2);
  if (start < 0) start = 0;
  if (start > Ld - Wd) start = Ld - Wd;

  // warm L2 for shared weights (global_prefetch_b8)
  __builtin_prefetch(Gt + tid * 36, 0, 1);
  __builtin_prefetch(WvWf + tid * 80, 0, 1);

  const float s0 = s0p[0];

  // ---- TDM: DMA the 256x80 f32 window tile into LDS staging ----
  if (wave == 0) {
    unsigned long long ga = (unsigned long long)(const void*)(x + start);
    unsigned int ldsOff   = (unsigned int)(unsigned long long)(void*)sStage;
    u32x4 g0;
    g0.x = 1u;                                        // count=1, user descriptor
    g0.y = ldsOff;                                    // lds_addr [63:32]
    g0.z = (unsigned int)(ga & 0xFFFFFFFFull);        // global_addr [95:64]
    g0.w = (unsigned int)((ga >> 32) & 0x01FFFFFFull) // global_addr [120:96]
         | (2u << 30);                                // type=2 ("image")
    i32x8 g1;
    g1.s0 = (2 << 16);                                // data_size = 4B
    g1.s1 = (int)(((unsigned)Ld & 0xFFFFu) << 16);    // tensor_dim0 lo16 @ [63:48]
    g1.s2 = (int)(((unsigned)Ld >> 16) | ((unsigned)Cd << 16)); // dim0 hi / dim1 lo
    g1.s3 = (int)(((unsigned)Cd >> 16) | ((unsigned)Wd << 16)); // dim1 hi / tile_dim0
    g1.s4 = Cd;                                       // tile_dim1 (tile_dim2 = 0)
    g1.s5 = Ld;                                       // tensor_dim0_stride lo32
    g1.s6 = 0;
    g1.s7 = 0;
    i32x4 z4 = {0, 0, 0, 0};
    i32x8 z8 = {0, 0, 0, 0, 0, 0, 0, 0};
    __builtin_amdgcn_tensor_load_to_lds(g0, g1, z4, z4, z8, 0);
    __builtin_amdgcn_s_wait_tensorcnt(0);
  }
  __syncthreads();

  // ---- convert staged window f32 -> f16 (row-major [c][w]), zero K-pad ----
  for (int i = 0; i < Wd; ++i) {
    int flat = i * 256 + tid;            // flat over c*80+w; stage layout matches
    int c = flat / Wd;
    int w = flat - c * Wd;
    sWin[c][w] = (_Float16)sStage[flat];
  }
  for (int i = tid; i < Cd * (Kp - Wd); i += 256) {
    int c = i >> 4, w = Wd + (i & 15);
    sWin[c][w] = (_Float16)0.f;
    sH[c][w]   = (_Float16)0.f;
  }
  __syncthreads();

  // ---- rank-1 terms: wa[c] = win[c,:].a , wb[c] = win[c,:].b ----
  {
    float aA = 0.f, aB = 0.f;
    for (int w = 0; w < Wd; ++w) {
      float wv = (float)sWin[tid][w];
      aA += wv * avec[w];
      aB += wv * bvec[w];
    }
    sWa[tid] = aA;
    sWb[tid] = aB;
  }

  // ---- H = win(256x96) * G(96x80): 2 m-tiles per wave, 5 n-tiles, 3 k ----
  for (int mi = wave * 2; mi < wave * 2 + 2; ++mi) {
    const int m0 = mi * 16;
    v16h a0 = loadA(&sWin[0][0], Kp, m0,  0, lane);
    v16h a1 = loadA(&sWin[0][0], Kp, m0, 32, lane);
    v16h a2 = loadA(&sWin[0][0], Kp, m0, 64, lane);
    for (int ni = 0; ni < 5; ++ni) {
      const int n0 = ni * 16;
      v16h b0 = loadBt(Gt, Kp, n0,  0, lane);
      v16h b1 = loadBt(Gt, Kp, n0, 32, lane);
      v16h b2 = loadBt(Gt, Kp, n0, 64, lane);
      v8f acc = {};
      acc = wmma32(a0, b0, acc);
      acc = wmma32(a1, b1, acc);
      acc = wmma32(a2, b2, acc);
      const int col   = n0 + (lane & 15);
      const int rbase = m0 + ((lane & 16) ? 8 : 0);
#pragma unroll
      for (int r = 0; r < 8; ++r) sH[rbase + r][col] = (_Float16)acc[r];
    }
  }
  __syncthreads();

  // ---- S = H(256x96) * win^T(96x256) + wa + wb + s0 ----
  for (int mi = wave * 2; mi < wave * 2 + 2; ++mi) {
    const int m0 = mi * 16;
    v16h a0 = loadA(&sH[0][0], Kp, m0,  0, lane);
    v16h a1 = loadA(&sH[0][0], Kp, m0, 32, lane);
    v16h a2 = loadA(&sH[0][0], Kp, m0, 64, lane);
    for (int ni = 0; ni < 16; ++ni) {
      const int n0 = ni * 16;
      v16h b0 = loadBt(&sWin[0][0], Kp, n0,  0, lane);
      v16h b1 = loadBt(&sWin[0][0], Kp, n0, 32, lane);
      v16h b2 = loadBt(&sWin[0][0], Kp, n0, 64, lane);
      v8f acc = {};
      acc = wmma32(a0, b0, acc);
      acc = wmma32(a1, b1, acc);
      acc = wmma32(a2, b2, acc);
      const int col   = n0 + (lane & 15);
      const int rbase = m0 + ((lane & 16) ? 8 : 0);
      const float cb  = sWb[col] + s0;
#pragma unroll
      for (int r = 0; r < 8; ++r)
        sS[rbase + r][col] = (_Float16)(acc[r] + sWa[rbase + r] + cb);
    }
  }
  __syncthreads();

  // ---- softmax row stats: thread = row, b128 loads, 8-wide exp ILP ----
  {
    const int c = tid;
    float m = -1e30f;
    for (int i = 0; i < Cd / 8; ++i) {
      int ch = ((i + tid) & (Cd / 8 - 1)) * 8;
      v8h v = *(const v8h*)&sS[c][ch];
#pragma unroll
      for (int j = 0; j < 8; ++j) m = fmaxf(m, (float)v[j]);
    }
    float z = 0.f;
    for (int i = 0; i < Cd / 8; ++i) {
      int ch = ((i + tid) & (Cd / 8 - 1)) * 8;
      v8h v = *(const v8h*)&sS[c][ch];
#pragma unroll
      for (int j = 0; j < 8; ++j) z += __expf((float)v[j] - m);
    }
    sRowMax[c]  = m;
    sRowInvZ[c] = 1.f / z;
  }
  __syncthreads();

  // ---- column means: wave owns 32 rows, lane owns 8 cols; then reduce ----
  {
    float acc[8];
#pragma unroll
    for (int j = 0; j < 8; ++j) acc[j] = 0.f;
    for (int i = 0; i < 32; ++i) {
      int c = wave * 32 + ((i + lane) & 31);
      float rm = sRowMax[c], rz = sRowInvZ[c];
      v8h v = *(const v8h*)&sS[c][lane * 8];
#pragma unroll
      for (int j = 0; j < 8; ++j) acc[j] += __expf((float)v[j] - rm) * rz;
    }
#pragma unroll
    for (int j = 0; j < 8; ++j) sPart[wave][lane * 8 + j] = acc[j];
  }
  __syncthreads();
  {
    float s = 0.f;
#pragma unroll
    for (int w8 = 0; w8 < 8; ++w8) s += sPart[w8][tid];
    sWcol[tid] = s * (1.0f / (float)Cd);
  }
  __syncthreads();

  // ---- u[j] = sum_d w[d] * win[d][j] ----
  if (tid < Kp) {
    float acc = 0.f;
    for (int d = 0; d < Cd; ++d) acc += sWcol[d] * (float)sWin[d][tid];
    sU[tid] = acc;
  }
  __syncthreads();

  // ---- out[c,l] = u . WvWf[:,c] + bvf[c] ----
  {
    const int c = tid;
    float acc = bvf[c];
    for (int j = 0; j < Wd; ++j) acc += sU[j] * WvWf[j * Cd + c];
    out[c * Ld + l] = acc;
  }
}

// =======================================================================
extern "C" void kernel_launch(void* const* d_in, const int* in_sizes, int n_in,
                              void* d_out, int out_size, void* d_ws, size_t ws_size,
                              hipStream_t stream) {
  const float* x  = (const float*)d_in[0];
  const float* Wq = (const float*)d_in[1];
  const float* bq = (const float*)d_in[2];
  const float* Wk = (const float*)d_in[3];
  const float* bk = (const float*)d_in[4];
  const float* Wv = (const float*)d_in[5];
  const float* bv = (const float*)d_in[6];
  const float* Wf = (const float*)d_in[7];
  const float* bf = (const float*)d_in[8];

  char* ws = (char*)d_ws;
  _Float16* Gt = (_Float16*)(ws + OFF_GT);
  float* avec  = (float*)(ws + OFF_AV);
  float* bvec  = (float*)(ws + OFF_BV);
  float* s0    = (float*)(ws + OFF_S0);
  float* WvWf  = (float*)(ws + OFF_WVWF);
  float* bvf   = (float*)(ws + OFF_BVF);

  const int preBlocks = (Kp*Kp + Kp + Kp + 1 + Wd*Cd + Cd + 255) / 256;
  la_precompute<<<preBlocks, 256, 0, stream>>>(Wq, bq, Wk, bk, Wv, bv, Wf, bf,
                                               Gt, avec, bvec, s0, WvWf, bvf);

  la_main<<<Ld, 256, 0, stream>>>(x, Gt, avec, bvec, s0, WvWf, bvf, (float*)d_out);
}
